// TanhAttention_28174985462282
// MI455X (gfx1250) — matrix operand order repaired and checked by reading, not compiled, not run on hardware
//
#include <hip/hip_runtime.h>
#include <math.h>

#define WAVE 32

typedef __attribute__((ext_vector_type(16))) __bf16 v16bf;
typedef __attribute__((ext_vector_type(8)))  __bf16 v8bf;
typedef __attribute__((ext_vector_type(8)))  float  v8f;

constexpr int B = 4, L = 256, T = 256, D = 512;

// ---- tanh: prefer the gfx1250 V_TANH_F32 hardware op -----------------------
#if defined(__HIP_DEVICE_COMPILE__)
# if __has_builtin(__builtin_amdgcn_tanhf)
#  define DEV_TANH(x) __builtin_amdgcn_tanhf(x)
# elif __has_builtin(__builtin_amdgcn_tanh_f32)
#  define DEV_TANH(x) __builtin_amdgcn_tanh_f32(x)
# else
#  define DEV_TANH(x) tanhf(x)
# endif
#else
# define DEV_TANH(x) tanhf(x)
#endif

__device__ __forceinline__ __bf16 bf_hi(float x) { return (__bf16)x; }
__device__ __forceinline__ __bf16 bf_lo(float x, __bf16 h) {
  return (__bf16)(x - (float)h);
}

__device__ __forceinline__ v16bf cat8(v8bf a, v8bf b) {
  return __builtin_shufflevector(a, b, 0, 1, 2, 3, 4, 5, 6, 7,
                                 8, 9, 10, 11, 12, 13, 14, 15);
}

// A tile: 16(M) x 32(K), bf16 layout per ISA:
//   lanes 0-15 : M=lane,    K = {0..7} then {16..23}
//   lanes 16-31: M=lane-16, K = {8..15} then {24..31}
__device__ __forceinline__ v16bf load_a_bf16(const __bf16* __restrict__ base,
                                             int ldk, int m0, int k0) {
  const int lane = threadIdx.x & (WAVE - 1);
  const __bf16* p = base + (size_t)(m0 + (lane & 15)) * ldk +
                    (k0 + ((lane >> 4) << 3));
  v8bf c0 = *(const v8bf*)p;          // 16B aligned: k chunk multiple of 8
  v8bf c1 = *(const v8bf*)(p + 16);
  return cat8(c0, c1);
}

// B tile: 32(K) x 16(N): lanes 0-15: N=lane, K = {0..7} then {16..23};
// lanes 16-31: K = {8..15} then {24..31}. Source is row-major B^T as [N rows x K].
__device__ __forceinline__ v16bf load_b_bf16(const __bf16* __restrict__ baseT,
                                             int ldk, int n0, int k0) {
  const int lane = threadIdx.x & (WAVE - 1);
  const __bf16* p = baseT + (size_t)(n0 + (lane & 15)) * ldk +
                    (k0 + ((lane >> 4) << 3));
  v8bf c0 = *(const v8bf*)p;
  v8bf c1 = *(const v8bf*)(p + 16);
  return cat8(c0, c1);
}

// 3-term split-bf16 product: Ah*Bh + Ah*Bl + Al*Bh (fp32 accumulate)
__device__ __forceinline__ v8f wmma3(v8f acc, v16bf ah, v16bf al,
                                     v16bf bh, v16bf bl) {
  acc = __builtin_amdgcn_wmma_f32_16x16x32_bf16(false, ah, false, bh,
                                                (short)0, acc, false, false);
  acc = __builtin_amdgcn_wmma_f32_16x16x32_bf16(false, ah, false, bl,
                                                (short)0, acc, false, false);
  acc = __builtin_amdgcn_wmma_f32_16x16x32_bf16(false, al, false, bh,
                                                (short)0, acc, false, false);
  return acc;
}

// ---- kernel 0: elementwise fp32 -> (hi, lo) bf16 split --------------------
__global__ void __launch_bounds__(256)
convsplit_kernel(const float* __restrict__ in, __bf16* __restrict__ hi,
                 __bf16* __restrict__ lo, int n) {
  const int i = blockIdx.x * 256 + threadIdx.x;
  if (i < n) {
    float x = in[i];
    __bf16 h = bf_hi(x);
    hi[i] = h;
    lo[i] = bf_lo(x, h);
  }
}

// ---- kernel 0b: memory[b,t,d] -> memT{hi,lo}[b,d,t] (transpose + split) ---
__global__ void __launch_bounds__(256)
transpose_split_kernel(const float* __restrict__ in, __bf16* __restrict__ hi,
                       __bf16* __restrict__ lo) {
  const int idx = blockIdx.x * 256 + threadIdx.x;   // over B*T*D
  const int d = idx % D;
  const int r = idx / D;
  const int t = r % T;
  const int b = r / T;
  const float x = in[idx];
  const __bf16 h = bf_hi(x);
  const size_t o = ((size_t)b * D + d) * T + t;
  hi[o] = h;
  lo[o] = bf_lo(x, h);
}

// ---- kernel 1: C[m,n] = sum_k A[m,k]*W[n,k] + bias[n]  (item1 / item2) ----
__global__ void __launch_bounds__(WAVE)
proj_wmma_kernel(const __bf16* __restrict__ Ahi, const __bf16* __restrict__ Alo,
                 const __bf16* __restrict__ Whi, const __bf16* __restrict__ Wlo,
                 const float* __restrict__ bias, float* __restrict__ Cmat,
                 int Ntot, int Ktot) {
  const int n0   = blockIdx.x * 16;
  const int m0   = blockIdx.y * 16;
  const int lane = threadIdx.x & (WAVE - 1);
  const float bv = bias[n0 + (lane & 15)];
  v8f acc = {bv, bv, bv, bv, bv, bv, bv, bv};
  for (int k0 = 0; k0 < Ktot; k0 += 32) {
    v16bf ah = load_a_bf16(Ahi, Ktot, m0, k0);
    v16bf al = load_a_bf16(Alo, Ktot, m0, k0);
    v16bf bh = load_b_bf16(Whi, Ktot, n0, k0);
    v16bf bl = load_b_bf16(Wlo, Ktot, n0, k0);
    acc = wmma3(acc, ah, al, bh, bl);
  }
  // C/D layout: element i -> M = i + 8*(lane>>4), N = lane&15
  const int mrow = m0 + ((lane >> 4) << 3);
  float* o = Cmat + (size_t)mrow * Ntot + n0 + (lane & 15);
#pragma unroll
  for (int i = 0; i < 8; ++i) o[(size_t)i * Ntot] = acc[i];
}

// ---- kernel 2: scores (tanh reduction) + mask + softmax -> P hi/lo --------
__global__ void __launch_bounds__(256)
score_softmax_kernel(const float* __restrict__ item1,
                     const float* __restrict__ item2,
                     const float* __restrict__ wt,
                     const float* __restrict__ btp,
                     const int*   __restrict__ mmask,
                     __bf16* __restrict__ Phi, __bf16* __restrict__ Plo) {
  __shared__ float s_i1[D];
  __shared__ float s_wt[D];
  __shared__ float red[256];
  const int bl = blockIdx.x;          // b*L + l
  const int b  = bl / L;
  const int l  = bl % L;
  const int t  = threadIdx.x;         // one score column per thread (T==256)

  const float* i1row = item1 + (size_t)bl * D;
  for (int d = t; d < D; d += 256) { s_i1[d] = i1row[d]; s_wt[d] = wt[d]; }
  __syncthreads();

  float S = -1e30f;
  const bool valid = (t >= l) && (mmask[b * T + t] != 0);
  if (valid) {                        // waves entirely below diagonal skip all work
    const float* i2row = item2 + ((size_t)b * T + t) * D;
    float acc = 0.0f;
#pragma unroll 8
    for (int d = 0; d < D; ++d)
      acc += DEV_TANH(s_i1[d] + i2row[d]) * s_wt[d];
    S = acc + btp[0];
  }

  // block softmax over 256 columns
  red[t] = S; __syncthreads();
  for (int off = 128; off > 0; off >>= 1) {
    if (t < off) red[t] = fmaxf(red[t], red[t + off]);
    __syncthreads();
  }
  const float m = red[0];
  __syncthreads();
  const float e = expf(S - m);        // fully-masked row -> e=1 everywhere -> uniform
  red[t] = e; __syncthreads();
  for (int off = 128; off > 0; off >>= 1) {
    if (t < off) red[t] += red[t + off];
    __syncthreads();
  }
  const float p = e / red[0];
  const size_t idx = (size_t)bl * T + t;
  const __bf16 h = bf_hi(p);
  Phi[idx] = h;
  Plo[idx] = bf_lo(p, h);
}

// ---- kernel 3: out[b] = P[b] @ memory[b]  via WMMA ------------------------
__global__ void __launch_bounds__(WAVE)
pv_wmma_kernel(const __bf16* __restrict__ Phi, const __bf16* __restrict__ Plo,
               const __bf16* __restrict__ MThi, const __bf16* __restrict__ MTlo,
               float* __restrict__ out) {
  const int n0   = blockIdx.x * 16;   // d
  const int m0   = blockIdx.y * 16;   // l
  const int b    = blockIdx.z;
  const int lane = threadIdx.x & (WAVE - 1);
  const __bf16* Ph = Phi + (size_t)b * L * T;
  const __bf16* Pl = Plo + (size_t)b * L * T;
  const __bf16* Mh = MThi + (size_t)b * D * T;
  const __bf16* Ml = MTlo + (size_t)b * D * T;
  v8f acc = {};
  for (int k0 = 0; k0 < T; k0 += 32) {
    v16bf ah = load_a_bf16(Ph, T, m0, k0);
    v16bf al = load_a_bf16(Pl, T, m0, k0);
    v16bf bh = load_b_bf16(Mh, T, n0, k0);
    v16bf bl = load_b_bf16(Ml, T, n0, k0);
    acc = wmma3(acc, ah, al, bh, bl);
  }
  const int mrow = m0 + ((lane >> 4) << 3);
  float* o = out + ((size_t)b * L + mrow) * D + n0 + (lane & 15);
#pragma unroll
  for (int i = 0; i < 8; ++i) o[(size_t)i * D] = acc[i];
}

extern "C" void kernel_launch(void* const* d_in, const int* in_sizes, int n_in,
                              void* d_out, int out_size, void* d_ws, size_t ws_size,
                              hipStream_t stream) {
  (void)in_sizes; (void)n_in; (void)out_size; (void)ws_size;
  const float* x    = (const float*)d_in[0];
  const float* mem  = (const float*)d_in[1];
  const float* W1   = (const float*)d_in[2];
  const float* b1   = (const float*)d_in[3];
  const float* W2   = (const float*)d_in[4];
  const float* b2   = (const float*)d_in[5];
  const float* wt   = (const float*)d_in[6];
  const float* bt   = (const float*)d_in[7];
  const int*   mask = (const int*)d_in[8];
  float* out = (float*)d_out;

  constexpr size_t BLD = (size_t)B * L * D;   // 524288
  constexpr size_t BTD = (size_t)B * T * D;   // 524288
  constexpr size_t DD  = (size_t)D * D;       // 262144
  constexpr size_t BLT = (size_t)B * L * T;   // 262144

  char* ws = (char*)d_ws;
  size_t off = 0;
  float*  item1 = (float*)(ws + off);  off += BLD * 4;
  float*  item2 = (float*)(ws + off);  off += BTD * 4;
  __bf16* xhi   = (__bf16*)(ws + off); off += BLD * 2;
  __bf16* xlo   = (__bf16*)(ws + off); off += BLD * 2;
  __bf16* mhi   = (__bf16*)(ws + off); off += BTD * 2;
  __bf16* mlo   = (__bf16*)(ws + off); off += BTD * 2;
  __bf16* w1hi  = (__bf16*)(ws + off); off += DD * 2;
  __bf16* w1lo  = (__bf16*)(ws + off); off += DD * 2;
  __bf16* w2hi  = (__bf16*)(ws + off); off += DD * 2;
  __bf16* w2lo  = (__bf16*)(ws + off); off += DD * 2;
  __bf16* mThi  = (__bf16*)(ws + off); off += BTD * 2;
  __bf16* mTlo  = (__bf16*)(ws + off); off += BTD * 2;
  __bf16* Phi   = (__bf16*)(ws + off); off += BLT * 2;
  __bf16* Plo   = (__bf16*)(ws + off); off += BLT * 2;

  // split every WMMA operand once (keeps GEMM loops conversion-free)
  convsplit_kernel<<<BLD / 256, 256, 0, stream>>>(x, xhi, xlo, (int)BLD);
  convsplit_kernel<<<BTD / 256, 256, 0, stream>>>(mem, mhi, mlo, (int)BTD);
  convsplit_kernel<<<DD / 256, 256, 0, stream>>>(W1, w1hi, w1lo, (int)DD);
  convsplit_kernel<<<DD / 256, 256, 0, stream>>>(W2, w2hi, w2lo, (int)DD);
  transpose_split_kernel<<<BTD / 256, 256, 0, stream>>>(mem, mThi, mTlo);

  dim3 gProj(D / 16, (B * L) / 16);   // (32, 64) tiles, 1 wave each
  proj_wmma_kernel<<<gProj, WAVE, 0, stream>>>(xhi, xlo, w1hi, w1lo, b1,
                                               item1, D, D);
  proj_wmma_kernel<<<gProj, WAVE, 0, stream>>>(mhi, mlo, w2hi, w2lo, b2,
                                               item2, D, D);
  score_softmax_kernel<<<B * L, 256, 0, stream>>>(item1, item2, wt, bt, mask,
                                                  Phi, Plo);
  pv_wmma_kernel<<<dim3(D / 16, L / 16, B), WAVE, 0, stream>>>(Phi, Plo,
                                                               mThi, mTlo, out);
}
